// MultiHeadAttention_28759101014445
// MI455X (gfx1250) — matrix operand order repaired and checked by reading, not compiled, not run on hardware
//
#include <hip/hip_runtime.h>
#include <hip/hip_bf16.h>

// Problem constants (match reference)
#define B_ 2
#define S_ 2048
#define E_ 1024
#define H_ 16
#define D_ 64
#define NEGINF (-1e9f)

typedef __bf16 bf16_t;
typedef __attribute__((ext_vector_type(16))) __bf16 v16bf;
typedef __attribute__((ext_vector_type(8)))  __bf16 v8bf;
typedef __attribute__((ext_vector_type(8)))  float  v8f;
typedef __attribute__((ext_vector_type(4)))  float  v4f;

__device__ __forceinline__ v8f wmma_bf16(v16bf a, v16bf b, v8f c) {
  // 8 args: (neg_a, A, neg_b, B, c_mod, C, reuse_a, reuse_b)
  return __builtin_amdgcn_wmma_f32_16x16x32_bf16(false, a, false, b, (short)0, c,
                                                 false, false);
}

// ---- WMMA operand loaders (layouts per CDNA5 ISA 7.12.2) ----
// A operand 16x32: lane L holds row m=L%16, half h=L/16.
//   elem e in [0,8):  K = k0 + 8h + e ; elem e in [8,16): K = k0 + 16 + 8h + (e-8)
__device__ __forceinline__ v16bf load_A_bf16(const bf16_t* p, int ld, int m, int k0, int h) {
  const bf16_t* r = p + (size_t)m * ld + k0;
  v8bf lo = *(const v8bf*)(r + 8 * h);
  v8bf hi = *(const v8bf*)(r + 16 + 8 * h);
  v16bf a;
#pragma unroll
  for (int i = 0; i < 8; i++) { a[i] = lo[i]; a[i + 8] = hi[i]; }
  return a;
}

// B operand 32x16: B[k][n] = src_row_n[k0 + k]; lane holds col n=L%16,
// half h=L/16, elem e<8 -> K = 8h + e ; e>=8 -> K = 16 + 8h + (e-8).
__device__ __forceinline__ v16bf load_B_bf16(const bf16_t* p, int ld, int n, int k0, int h) {
  const bf16_t* r = p + (size_t)n * ld + k0;
  v8bf lo = *(const v8bf*)(r + 8 * h);
  v8bf hi = *(const v8bf*)(r + 16 + 8 * h);
  v16bf b;
#pragma unroll
  for (int i = 0; i < 8; i++) { b[i] = lo[i]; b[i + 8] = hi[i]; }
  return b;
}

// ============ Kernel 0: elementwise f32 -> bf16 (weights & activations) ============
__global__ void __launch_bounds__(256)
cvt_bf16_kernel(const float* __restrict__ src, bf16_t* __restrict__ dst, int n8) {
  const int i = blockIdx.x * 256 + threadIdx.x;
  if (i >= n8) return;
  v4f a = *(const v4f*)(src + (size_t)i * 8);
  v4f b = *(const v4f*)(src + (size_t)i * 8 + 4);
  v8bf o;
#pragma unroll
  for (int j = 0; j < 4; j++) { o[j] = (__bf16)a[j]; o[j + 4] = (__bf16)b[j]; }
  *(v8bf*)(dst + (size_t)i * 8) = o;
}

// ============ Kernel 1: input projection  Y = X @ W^T + bias (all-bf16 loads) ======
// X: [B*S, E] bf16, W: [E, E] bf16 (used transposed). 32x64 output tile per wave.
// mode 0: store [B,H,S,D] bf16 ; mode 1: store transposed [B,H,D,S] bf16 (for V).
__global__ void __launch_bounds__(32)
proj_kernel(const bf16_t* __restrict__ X, const bf16_t* __restrict__ W,
            const float* __restrict__ bias, bf16_t* __restrict__ Y, int mode) {
  const int lane = threadIdx.x;
  const int h    = lane >> 4;
  const int nl   = lane & 15;
  const int mb   = blockIdx.x * 32;   // row tile in [0, B*S)
  const int nb   = blockIdx.y * 64;   // col tile in [0, E)

  v8f acc[2][4] = {};
  for (int k0 = 0; k0 < E_; k0 += 32) {
    v16bf a0 = load_A_bf16(X, E_, mb + nl, k0, h);
    v16bf a1 = load_A_bf16(X, E_, mb + 16 + nl, k0, h);
#pragma unroll
    for (int t = 0; t < 4; t++) {
      v16bf bm = load_B_bf16(W, E_, nb + t * 16 + nl, k0, h);
      acc[0][t] = wmma_bf16(a0, bm, acc[0][t]);
      acc[1][t] = wmma_bf16(a1, bm, acc[1][t]);
    }
  }
#pragma unroll
  for (int t = 0; t < 4; t++) {
    const int ncol = nb + t * 16 + nl;
    const float bv = bias[ncol];
    const int hh = ncol / D_, d = ncol % D_;
#pragma unroll
    for (int mt = 0; mt < 2; mt++)
#pragma unroll
      for (int r = 0; r < 8; r++) {
        const int m = mb + mt * 16 + r + 8 * h;   // C layout: M = r + 8h
        const int b = m / S_, s = m % S_;
        const float val = acc[mt][t][r] + bv;
        size_t idx;
        if (mode == 0) idx = (((size_t)(b * H_ + hh)) * S_ + s) * D_ + d;
        else           idx = (((size_t)(b * H_ + hh)) * D_ + d) * S_ + s;
        Y[idx] = (bf16_t)val;
      }
  }
}

// ============ Kernel 2: causal flash attention per (head, 16-query tile) ============
// Qb,Kb: [B,H,S,D] bf16. Vt: [B,H,D,S] bf16. Ctx out: [B*S, E] bf16.
// K tiles (32 keys x 64 = 4KB, contiguous) are double-buffered in LDS via the
// gfx1250 async-to-LDS path (global_load_async_to_lds_b128 + s_wait_asynccnt).
__device__ __forceinline__ void async_load_ktile(const bf16_t* Kp, int kb,
                                                 unsigned lds_lane, int lane) {
  unsigned long long g = (unsigned long long)(size_t)Kp
                       + (unsigned long long)kb * (D_ * 2) + (unsigned)lane * 16u;
  // 8 x 512B slabs cover the 4KB tile; offset applies to both LDS and global addr.
  asm volatile("global_load_async_to_lds_b128 %0, %1, off"             :: "v"(lds_lane), "v"(g) : "memory");
  asm volatile("global_load_async_to_lds_b128 %0, %1, off offset:512"  :: "v"(lds_lane), "v"(g) : "memory");
  asm volatile("global_load_async_to_lds_b128 %0, %1, off offset:1024" :: "v"(lds_lane), "v"(g) : "memory");
  asm volatile("global_load_async_to_lds_b128 %0, %1, off offset:1536" :: "v"(lds_lane), "v"(g) : "memory");
  asm volatile("global_load_async_to_lds_b128 %0, %1, off offset:2048" :: "v"(lds_lane), "v"(g) : "memory");
  asm volatile("global_load_async_to_lds_b128 %0, %1, off offset:2560" :: "v"(lds_lane), "v"(g) : "memory");
  asm volatile("global_load_async_to_lds_b128 %0, %1, off offset:3072" :: "v"(lds_lane), "v"(g) : "memory");
  asm volatile("global_load_async_to_lds_b128 %0, %1, off offset:3584" :: "v"(lds_lane), "v"(g) : "memory");
}

__global__ void __launch_bounds__(32)
attn_kernel(const bf16_t* __restrict__ Qb, const bf16_t* __restrict__ Kb,
            const bf16_t* __restrict__ Vt, const unsigned char* __restrict__ pad,
            bf16_t* __restrict__ Ctx) {
  __shared__ bf16_t Kbuf[2][32 * D_];   // 2 x 4KB K-tile double buffer
  __shared__ bf16_t Plds[16 * 32];      // P tile staging (C-layout -> A-layout)

  const int lane = threadIdx.x;
  const int h    = lane >> 4;
  const int nl   = lane & 15;
  const int qb   = blockIdx.x * 16;     // query tile base
  const int bh   = blockIdx.y;          // b*H + h
  const int b    = bh / H_;
  const int hh   = bh % H_;

  const bf16_t* Q  = Qb + (size_t)bh * S_ * D_;
  const bf16_t* Kp = Kb + (size_t)bh * S_ * D_;
  const bf16_t* Vp = Vt + (size_t)bh * D_ * S_;

  const unsigned lds_lane0 = (unsigned)(size_t)&Kbuf[0][0] + (unsigned)lane * 16u;
  const unsigned lds_lane1 = (unsigned)(size_t)&Kbuf[1][0] + (unsigned)lane * 16u;

  // Q A-operands for both 32-wide D chunks, loaded once.
  const v16bf a0 = load_A_bf16(Q, D_, qb + nl, 0, h);
  const v16bf a1 = load_A_bf16(Q, D_, qb + nl, 32, h);

  float rmax[8], rsum[8];
#pragma unroll
  for (int r = 0; r < 8; r++) { rmax[r] = -1e30f; rsum[r] = 0.f; }
  v8f acc[4] = {};                      // ctx accumulator, 16x64
  const float scale = 0.125f;           // 1/sqrt(D)

  const int nchunk = (qb >> 5) + 1;     // chunks of 32 keys covering [0, qb+16)
  async_load_ktile(Kp, 0, lds_lane0, lane);   // prologue prefetch

  for (int ci = 0; ci < nchunk; ci++) {
    const int kb = ci * 32;
    // Prefetch next chunk into other buffer, then wait for current tile only.
    if (ci + 1 < nchunk) {
      async_load_ktile(Kp, kb + 32, (ci & 1) ? lds_lane0 : lds_lane1, lane);
      asm volatile("s_wait_asynccnt 8" ::: "memory");
    } else {
      asm volatile("s_wait_asynccnt 0" ::: "memory");
    }
    const bf16_t* kt = &Kbuf[ci & 1][0];

    // ---- scores: two 16x16 tiles (keys kb..kb+15, kb+16..kb+31), K from LDS ----
    v8f sc[2] = {};
#pragma unroll
    for (int t = 0; t < 2; t++) {
      const int row = t * 16 + nl;      // key within tile
      v16bf b0 = load_B_bf16(kt, D_, row, 0, h);
      v16bf b1 = load_B_bf16(kt, D_, row, 32, h);
      sc[t] = wmma_bf16(a0, b0, sc[t]);
      sc[t] = wmma_bf16(a1, b1, sc[t]);
    }
    // ---- mask + scale (C layout: row q = qb + r + 8h, col key = kb + 16t + nl) ----
#pragma unroll
    for (int t = 0; t < 2; t++) {
      const int key = kb + t * 16 + nl;
      const float pm = pad[b * S_ + key] ? NEGINF : 0.f;
#pragma unroll
      for (int r = 0; r < 8; r++) {
        const int q = qb + r + 8 * h;
        float v = sc[t][r] * scale + pm;
        if (key > q) v += NEGINF;       // causal_mask adds NEG
        sc[t][r] = v;
      }
    }
    // ---- online softmax (row reductions across 16 lanes of each half) ----
#pragma unroll
    for (int r = 0; r < 8; r++) {
      float mx = fmaxf(sc[0][r], sc[1][r]);
#pragma unroll
      for (int off = 8; off >= 1; off >>= 1)
        mx = fmaxf(mx, __shfl_xor(mx, off, 16));
      const float nm = fmaxf(rmax[r], mx);
      const float corr = __expf(rmax[r] - nm);
      rmax[r] = nm;
      rsum[r] *= corr;
#pragma unroll
      for (int t = 0; t < 4; t++) acc[t][r] *= corr;
      const float e0 = __expf(sc[0][r] - nm);
      const float e1 = __expf(sc[1][r] - nm);
      sc[0][r] = e0; sc[1][r] = e1;
      float sum = e0 + e1;
#pragma unroll
      for (int off = 8; off >= 1; off >>= 1)
        sum += __shfl_xor(sum, off, 16);
      rsum[r] += sum;
    }
    // ---- P (16x32, C layout) -> LDS -> A-layout operand ----
#pragma unroll
    for (int t = 0; t < 2; t++)
#pragma unroll
      for (int r = 0; r < 8; r++)
        Plds[(r + 8 * h) * 32 + t * 16 + nl] = (bf16_t)sc[t][r];
    asm volatile("s_wait_dscnt 0" ::: "memory");   // LDS RAW within wave
    const v16bf pA = load_A_bf16(Plds, 32, nl, 0, h);
    // ---- ctx += P @ V  (B operand from transposed V: contiguous along keys) ----
#pragma unroll
    for (int t = 0; t < 4; t++) {
      v16bf bv = load_B_bf16(Vp, S_, t * 16 + nl, kb, h);
      acc[t] = wmma_bf16(pA, bv, acc[t]);
    }
  }

  // ---- finalize: divide by row sums, store ctx tile [16 x 64] ----
#pragma unroll
  for (int t = 0; t < 4; t++)
#pragma unroll
    for (int r = 0; r < 8; r++) {
      const int q = qb + r + 8 * h;
      const int d = t * 16 + nl;
      const float val = acc[t][r] / rsum[r];
      Ctx[((size_t)(b * S_ + q)) * E_ + hh * D_ + d] = (bf16_t)val;
    }
}

// ============ Kernel 3: output projection  out = Ctx @ Wo^T + bo (f32 out) =========
__global__ void __launch_bounds__(32)
outproj_kernel(const bf16_t* __restrict__ Ctx, const bf16_t* __restrict__ Wo,
               const float* __restrict__ bo, float* __restrict__ out) {
  const int lane = threadIdx.x;
  const int h    = lane >> 4;
  const int nl   = lane & 15;
  const int mb   = blockIdx.x * 32;
  const int nb   = blockIdx.y * 64;

  v8f acc[2][4] = {};
  for (int k0 = 0; k0 < E_; k0 += 32) {
    v16bf a0 = load_A_bf16(Ctx, E_, mb + nl, k0, h);
    v16bf a1 = load_A_bf16(Ctx, E_, mb + 16 + nl, k0, h);
#pragma unroll
    for (int t = 0; t < 4; t++) {
      v16bf bm = load_B_bf16(Wo, E_, nb + t * 16 + nl, k0, h);
      acc[0][t] = wmma_bf16(a0, bm, acc[0][t]);
      acc[1][t] = wmma_bf16(a1, bm, acc[1][t]);
    }
  }
#pragma unroll
  for (int t = 0; t < 4; t++) {
    const int ncol = nb + t * 16 + nl;
    const float bv = bo[ncol];
#pragma unroll
    for (int mt = 0; mt < 2; mt++)
#pragma unroll
      for (int r = 0; r < 8; r++) {
        const int m = mb + mt * 16 + r + 8 * h;
        out[(size_t)m * E_ + ncol] = acc[mt][t][r] + bv;
      }
  }
}

extern "C" void kernel_launch(void* const* d_in, const int* in_sizes, int n_in,
                              void* d_out, int out_size, void* d_ws, size_t ws_size,
                              hipStream_t stream) {
  (void)in_sizes; (void)n_in; (void)out_size; (void)ws_size;
  const float* Qin = (const float*)d_in[0];
  const float* Kin = (const float*)d_in[1];
  const float* Vin = (const float*)d_in[2];
  // d_in[3] = causal_mask (applied analytically), d_in[4] = padding_mask (bool)
  const unsigned char* pad = (const unsigned char*)d_in[4];
  const float* Wq = (const float*)d_in[5];  const float* bq = (const float*)d_in[6];
  const float* Wk = (const float*)d_in[7];  const float* bk = (const float*)d_in[8];
  const float* Wv = (const float*)d_in[9];  const float* bv = (const float*)d_in[10];
  const float* Wo = (const float*)d_in[11]; const float* bo = (const float*)d_in[12];

  const size_t per = (size_t)B_ * S_ * E_;     // 4Mi elements
  const size_t wsz = (size_t)E_ * E_;          // 1Mi elements
  bf16_t* Qb  = (bf16_t*)d_ws;                 // [B,H,S,D]
  bf16_t* Kb  = Qb + per;                      // [B,H,S,D]
  bf16_t* Vt  = Kb + per;                      // [B,H,D,S]
  bf16_t* Ctx = Vt + per;                      // [B*S, E]
  bf16_t* Wqb = Ctx + per;
  bf16_t* Wkb = Wqb + wsz;
  bf16_t* Wvb = Wkb + wsz;
  bf16_t* Wob = Wvb + wsz;
  bf16_t* Xq  = Wob + wsz;                     // bf16 copies of activations
  bf16_t* Xk  = Xq + per;
  bf16_t* Xv  = Xk + per;                      // total: 64 MB

  const dim3 blk(32);                          // one wave32 per block
  const int w8 = (int)(wsz / 8), p8 = (int)(per / 8);
  cvt_bf16_kernel<<<dim3(w8 / 256), 256, 0, stream>>>(Wq, Wqb, w8);
  cvt_bf16_kernel<<<dim3(w8 / 256), 256, 0, stream>>>(Wk, Wkb, w8);
  cvt_bf16_kernel<<<dim3(w8 / 256), 256, 0, stream>>>(Wv, Wvb, w8);
  cvt_bf16_kernel<<<dim3(w8 / 256), 256, 0, stream>>>(Wo, Wob, w8);
  cvt_bf16_kernel<<<dim3(p8 / 256), 256, 0, stream>>>(Qin, Xq, p8);
  cvt_bf16_kernel<<<dim3(p8 / 256), 256, 0, stream>>>(Kin, Xk, p8);
  cvt_bf16_kernel<<<dim3(p8 / 256), 256, 0, stream>>>(Vin, Xv, p8);

  const dim3 gProj(B_ * S_ / 32, E_ / 64);     // (128, 16)
  proj_kernel<<<gProj, blk, 0, stream>>>(Xq, Wqb, bq, Qb, 0);
  proj_kernel<<<gProj, blk, 0, stream>>>(Xk, Wkb, bk, Kb, 0);
  proj_kernel<<<gProj, blk, 0, stream>>>(Xv, Wvb, bv, Vt, 1);   // V transposed

  const dim3 gAttn(S_ / 16, B_ * H_);          // (128, 32)
  attn_kernel<<<gAttn, blk, 0, stream>>>(Qb, Kb, Vt, pad, Ctx);

  outproj_kernel<<<gProj, blk, 0, stream>>>(Ctx, Wob, bo, (float*)d_out);
}
